// SelfAttentionLayer_25993142075600
// MI455X (gfx1250) — compile-verified, matrix operand order checked by reading, not run on hardware
//
#include <hip/hip_runtime.h>
#include <hip/hip_bf16.h>

typedef __attribute__((ext_vector_type(16))) _Float16 v16h;
typedef __attribute__((ext_vector_type(8)))  _Float16 v8h;
typedef __attribute__((ext_vector_type(8)))  float    v8f;
typedef __attribute__((ext_vector_type(4)))  int      v4i;

#define B_  16
#define T_  4096
#define D_  256
#define N_  256
#define LAB 64
#define INP 96
#define MED 96

// One 256x256 f16 matrix in B-fragment layout:
// [kc(8)][nt(16)][lane(32)][h(16)] halfs
#define FRAG_HALFS (8 * 16 * 32 * 16)   // 65536

#if __has_builtin(__builtin_amdgcn_global_load_async_to_lds_b128) && \
    __has_builtin(__builtin_amdgcn_s_wait_asynccnt)
#define HAVE_ASYNC 1
typedef __attribute__((address_space(1))) v4i* as1_v4i;
typedef __attribute__((address_space(3))) v4i* as3_v4i;
// Copy 16 bytes global -> LDS via the async path (ASYNCcnt tracked).
__device__ __forceinline__ void async_copy16(const _Float16* g, _Float16* l) {
  __builtin_amdgcn_global_load_async_to_lds_b128((as1_v4i)g, (as3_v4i)l, 0, 0);
}
#else
#define HAVE_ASYNC 0
#endif

// 16-bit WMMA operand K-striping per CDNA5 ISA:
// lanes 0-15 get K {0..7, 16..23}, lanes 16-31 get K {8..15, 24..31}
__device__ __forceinline__ int koff(int lane, int h) {
  return (h & 7) + ((h & 8) << 1) + ((lane & 16) >> 1);
}

__device__ __forceinline__ v8f wmma16(v16h a, v16h b, v8f c) {
  return __builtin_amdgcn_wmma_f32_16x16x32_f16(false, a, false, b, (short)0, c,
                                                false, false);
}

// Load one f16 A-fragment (16x32 K-chunk kc) from an f32 row in global memory.
__device__ __forceinline__ v16h load_a_f32row(const float* __restrict__ row,
                                              int kc, int lane) {
  const int off0 = kc * 32 + ((lane & 16) >> 1);
  const float4 p0 = *(const float4*)(row + off0);
  const float4 p1 = *(const float4*)(row + off0 + 4);
  const float4 p2 = *(const float4*)(row + off0 + 16);
  const float4 p3 = *(const float4*)(row + off0 + 20);
  v16h a;
  a[0]  = (_Float16)p0.x; a[1]  = (_Float16)p0.y;
  a[2]  = (_Float16)p0.z; a[3]  = (_Float16)p0.w;
  a[4]  = (_Float16)p1.x; a[5]  = (_Float16)p1.y;
  a[6]  = (_Float16)p1.z; a[7]  = (_Float16)p1.w;
  a[8]  = (_Float16)p2.x; a[9]  = (_Float16)p2.y;
  a[10] = (_Float16)p2.z; a[11] = (_Float16)p2.w;
  a[12] = (_Float16)p3.x; a[13] = (_Float16)p3.y;
  a[14] = (_Float16)p3.z; a[15] = (_Float16)p3.w;
  return a;
}

// Load one f16 A-fragment from a row-major f16 row in LDS: the koff pattern is
// two contiguous 16B chunks -> 2x ds_load_b128.
__device__ __forceinline__ v16h lds_frag_row(const _Float16* __restrict__ rowbase,
                                             int kc, int lane) {
  const int off0 = kc * 32 + ((lane & 16) >> 1);
  const v8h lo = *(const v8h*)(rowbase + off0);
  const v8h hi = *(const v8h*)(rowbase + off0 + 16);
  return __builtin_shufflevector(lo, hi, 0, 1, 2, 3, 4, 5, 6, 7,
                                 8, 9, 10, 11, 12, 13, 14, 15);
}

__device__ __forceinline__ const float* hc_row(const float* __restrict__ h_l,
                                               const float* __restrict__ h_i,
                                               const float* __restrict__ h_m,
                                               int b, int n) {
  if (n < LAB)             return h_l + ((size_t)b * LAB + n) * D_;
  else if (n < LAB + INP)  return h_i + ((size_t)b * INP + (n - LAB)) * D_;
  else                     return h_m + ((size_t)b * MED + (n - LAB - INP)) * D_;
}

// ---------------------------------------------------------------------------
// Kernel 0: W^T (B[k][n] = W[n][k]) -> f16 fragment layout in workspace.
// ---------------------------------------------------------------------------
__global__ __launch_bounds__(32)
void swizzle_weights(const float* __restrict__ Wq, const float* __restrict__ Wk,
                     const float* __restrict__ Wv, _Float16* __restrict__ ws) {
  const int lane = threadIdx.x & 31;
  const int f = blockIdx.x;                 // fragment id: kc*16 + nt
  const int w = blockIdx.y;
  const int kc = f >> 4;
  const int nt = f & 15;
  const float* W = (w == 0) ? Wq : (w == 1) ? Wk : Wv;
  _Float16* dst = ws + (size_t)w * FRAG_HALFS;
  const int n = nt * 16 + (lane & 15);
  const float* row = W + (size_t)n * D_;    // B[k][n] = W[n][k]
  v16h o = load_a_f32row(row, kc, lane);    // same koff gather pattern
  *(v16h*)(dst + ((size_t)(f * 32 + lane)) * 16) = o;
}

// ---------------------------------------------------------------------------
// Kernel 1: K = h_c Wk^T, V = h_c Wv^T via WMMA.
//   - writes pass-through outputs h_{l,i,m} + V (f32)
//   - stores K^T fragments and V fragments (f16) for the attention kernel
// ---------------------------------------------------------------------------
__global__ __launch_bounds__(32)
void proj_kv(const float* __restrict__ h_l, const float* __restrict__ h_i,
             const float* __restrict__ h_m, _Float16* __restrict__ ws,
             float* __restrict__ out) {
  const int lane = threadIdx.x & 31;
  const int b   = blockIdx.x >> 3;
  const int kcv = blockIdx.x & 7;           // n-chunk: rows n0..n0+31
  const int n0  = kcv * 32;

  __shared__ alignas(32) _Float16 lds_k[32][D_];
  __shared__ alignas(32) _Float16 lds_v[32][D_];

  v16h a[2][8];
#pragma unroll
  for (int mt = 0; mt < 2; ++mt) {
    const int n = n0 + mt * 16 + (lane & 15);
    const float* row = hc_row(h_l, h_i, h_m, b, n);
#pragma unroll
    for (int kc = 0; kc < 8; ++kc) a[mt][kc] = load_a_f32row(row, kc, lane);
  }

  const _Float16* wkf = ws + (size_t)1 * FRAG_HALFS;
  const _Float16* wvf = ws + (size_t)2 * FRAG_HALFS;
  _Float16* ktf = ws + (size_t)(3 + b) * FRAG_HALFS;
  _Float16* vf  = ws + (size_t)(19 + b) * FRAG_HALFS;

  const int colb = lane & 15;
  const int rb   = (lane & 16) >> 1;

  // ---- K = h_c @ Wk^T ----
  for (int et = 0; et < 16; ++et) {
    v8f acc0 = {}; v8f acc1 = {};
#pragma unroll
    for (int kc = 0; kc < 8; ++kc) {
      v16h bf = *(const v16h*)(wkf + ((size_t)((kc * 16 + et) * 32 + lane)) * 16);
      acc0 = wmma16(a[0][kc], bf, acc0);
      acc1 = wmma16(a[1][kc], bf, acc1);
    }
    const int col = et * 16 + colb;
#pragma unroll
    for (int j = 0; j < 8; ++j) {
      lds_k[j + rb][col]      = (_Float16)acc0[j];
      lds_k[16 + j + rb][col] = (_Float16)acc1[j];
    }
  }

  // ---- V = h_c @ Wv^T (+ f32 pass-through outputs) ----
  float* out_l = out + (size_t)B_ * T_ * D_;
  float* out_i = out_l + (size_t)B_ * LAB * D_;
  float* out_m = out_i + (size_t)B_ * INP * D_;
  for (int et = 0; et < 16; ++et) {
    v8f acc0 = {}; v8f acc1 = {};
#pragma unroll
    for (int kc = 0; kc < 8; ++kc) {
      v16h bf = *(const v16h*)(wvf + ((size_t)((kc * 16 + et) * 32 + lane)) * 16);
      acc0 = wmma16(a[0][kc], bf, acc0);
      acc1 = wmma16(a[1][kc], bf, acc1);
    }
    const int col = et * 16 + colb;
#pragma unroll
    for (int j = 0; j < 8; ++j) {
#pragma unroll
      for (int half = 0; half < 2; ++half) {
        const float v = half ? acc1[j] : acc0[j];
        const int rowl = half * 16 + j + rb;
        lds_v[rowl][col] = (_Float16)v;
        const int ng = n0 + rowl;
        if (ng < LAB) {
          const size_t idx = ((size_t)b * LAB + ng) * D_ + col;
          out_l[idx] = h_l[idx] + v;
        } else if (ng < LAB + INP) {
          const size_t idx = ((size_t)b * INP + (ng - LAB)) * D_ + col;
          out_i[idx] = h_i[idx] + v;
        } else {
          const size_t idx = ((size_t)b * MED + (ng - LAB - INP)) * D_ + col;
          out_m[idx] = h_m[idx] + v;
        }
      }
    }
  }
  __syncthreads();

  // ---- K^T fragments: B[k=e][n] (row-major LDS gather -> 2x b128 each) ----
#pragma unroll
  for (int ntl = 0; ntl < 2; ++ntl) {
    const int rowl = ntl * 16 + (lane & 15);
    for (int kce = 0; kce < 8; ++kce) {
      v16h o = lds_frag_row(&lds_k[rowl][0], kce, lane);
      *(v16h*)(ktf +
               ((size_t)((kce * 16 + (2 * kcv + ntl)) * 32 + lane)) * 16) = o;
    }
  }
  // ---- V fragments: B[k=n][n=e] (column gather, scalar) ----
  for (int et = 0; et < 16; ++et) {
    const int col = et * 16 + (lane & 15);
    v16h o;
#pragma unroll
    for (int h = 0; h < 16; ++h) o[h] = lds_v[koff(lane, h)][col];
    *(v16h*)(vf + ((size_t)((kcv * 16 + et) * 32 + lane)) * 16) = o;
  }
}

// ---------------------------------------------------------------------------
// Block-cooperative 16x256 x 256x256 tile matmul.
// B fragments are staged once per block into LDS (double-buffered, async when
// available) and consumed by all 8 waves. accs[et] = A(16x256) @ B(:,et-tile).
// ---------------------------------------------------------------------------
__device__ __forceinline__ void mm16(const _Float16* __restrict__ fragbase,
                                     const v16h a[8], v8f accs[16],
                                     _Float16* __restrict__ lbuf,
                                     int tid, int lane) {
  const int lkc   = tid >> 5;   // fragment kc this thread stages (0..7)
  const int llane = tid & 31;
  const int loff  = (lkc * 32 + llane) * 16;

  // prologue: stage et=0 into buffer 0
  {
    const _Float16* g = fragbase + ((size_t)((lkc * 16 + 0) * 32 + llane)) * 16;
#if HAVE_ASYNC
    async_copy16(g, lbuf + loff);
    async_copy16(g + 8, lbuf + loff + 8);
    __builtin_amdgcn_s_wait_asynccnt(0);
#else
    *(v16h*)(lbuf + loff) = *(const v16h*)g;
#endif
  }
  __syncthreads();

#pragma unroll
  for (int et = 0; et < 16; ++et) {
    const int cur = (et & 1) * 4096;
    const int nxt = ((et + 1) & 1) * 4096;
#if HAVE_ASYNC
    if (et < 15) {
      const _Float16* g =
          fragbase + ((size_t)((lkc * 16 + et + 1) * 32 + llane)) * 16;
      async_copy16(g, lbuf + nxt + loff);
      async_copy16(g + 8, lbuf + nxt + loff + 8);
    }
#else
    v16h pre;
    if (et < 15)
      pre = *(const v16h*)(fragbase +
                           ((size_t)((lkc * 16 + et + 1) * 32 + llane)) * 16);
#endif
    v8f acc = {};
#pragma unroll
    for (int kc = 0; kc < 8; ++kc) {
      v16h bf = *(const v16h*)(lbuf + cur + (kc * 32 + lane) * 16);
      acc = wmma16(a[kc], bf, acc);
    }
    accs[et] = acc;
#if HAVE_ASYNC
    __builtin_amdgcn_s_wait_asynccnt(0);
#else
    if (et < 15) *(v16h*)(lbuf + nxt + loff) = pre;
#endif
    __syncthreads();
  }
}

// ---------------------------------------------------------------------------
// Kernel 2: attention. 256 threads = 8 waves; block covers 128 rows of T.
// Q = h_e Wq^T ; S = Q K^T / 16 ; P = softmax(S) (in registers) ; Z = P V.
// ---------------------------------------------------------------------------
__global__ __launch_bounds__(256)
void attention(const float* __restrict__ h_e, const _Float16* __restrict__ ws,
               const int* __restrict__ length, float* __restrict__ out) {
  const int tid  = threadIdx.x;
  const int lane = tid & 31;
  const int wave = tid >> 5;
  const int b  = blockIdx.x >> 5;                       // 32 blocks per batch
  const int t0 = (blockIdx.x & 31) * 128 + wave * 16;   // this wave's 16 rows

  __shared__ alignas(32) _Float16 lbuf[2 * 4096];       // 16 KB frag staging
  __shared__ alignas(32) _Float16 s_p[8][16 * 256];     // 8 KB per wave
  _Float16* sp = s_p[wave];

  const _Float16* wqf = ws;
  const _Float16* ktf = ws + (size_t)(3 + b) * FRAG_HALFS;
  const _Float16* vf  = ws + (size_t)(19 + b) * FRAG_HALFS;

  const int colb = lane & 15;
  const int rb   = (lane & 16) >> 1;

  // h_e A fragments (16 rows x 256)
  const float* arow = h_e + ((size_t)b * T_ + t0 + (lane & 15)) * D_;
  v16h ah[8];
#pragma unroll
  for (int kc = 0; kc < 8; ++kc) ah[kc] = load_a_f32row(arow, kc, lane);

  v8f accs[16];

  // ---- Q = A @ Wq^T ----
  mm16(wqf, ah, accs, lbuf, tid, lane);
#pragma unroll
  for (int et = 0; et < 16; ++et)
#pragma unroll
    for (int j = 0; j < 8; ++j)
      sp[(j + rb) * 256 + et * 16 + colb] = (_Float16)accs[et][j];

  // re-read Q in A-fragment layout (wave-local, DS ops are in order)
  v16h qf[8];
#pragma unroll
  for (int kc = 0; kc < 8; ++kc)
    qf[kc] = lds_frag_row(sp + (lane & 15) * 256, kc, lane);

  // ---- scores = Q @ K^T ----
  mm16(ktf, qf, accs, lbuf, tid, lane);

  // ---- softmax in registers; row r = j + rb lives in one 16-lane half ----
#pragma unroll
  for (int nt = 0; nt < 16; ++nt)
#pragma unroll
    for (int j = 0; j < 8; ++j) accs[nt][j] *= 0.0625f;   // 1/sqrt(256)

  float m[8], sum[8];
#pragma unroll
  for (int j = 0; j < 8; ++j) {
    float mm = -3.0e38f;
#pragma unroll
    for (int nt = 0; nt < 16; ++nt) mm = fmaxf(mm, accs[nt][j]);
#pragma unroll
    for (int mask = 1; mask < 16; mask <<= 1)
      mm = fmaxf(mm, __shfl_xor(mm, mask, 32));
    m[j] = mm;
  }
#pragma unroll
  for (int nt = 0; nt < 16; ++nt)
#pragma unroll
    for (int j = 0; j < 8; ++j) accs[nt][j] = __expf(accs[nt][j] - m[j]);
#pragma unroll
  for (int j = 0; j < 8; ++j) {
    float s = 0.0f;
#pragma unroll
    for (int nt = 0; nt < 16; ++nt) s += accs[nt][j];
#pragma unroll
    for (int mask = 1; mask < 16; mask <<= 1) s += __shfl_xor(s, mask, 32);
    sum[j] = 1.0f / s;
  }
#pragma unroll
  for (int nt = 0; nt < 16; ++nt)
#pragma unroll
    for (int j = 0; j < 8; ++j)
      sp[(j + rb) * 256 + nt * 16 + colb] = (_Float16)(accs[nt][j] * sum[j]);

  // probs in A-fragment layout
  v16h pf[8];
#pragma unroll
  for (int nc = 0; nc < 8; ++nc)
    pf[nc] = lds_frag_row(sp + (lane & 15) * 256, nc, lane);

  // ---- Z = P @ V ----
  mm16(vf, pf, accs, lbuf, tid, lane);

  const int len = length[b];
#pragma unroll
  for (int et = 0; et < 16; ++et) {
#pragma unroll
    for (int j = 0; j < 8; ++j) {
      const int t = t0 + j + rb;
      const size_t idx = ((size_t)b * T_ + t) * D_ + et * 16 + colb;
      const float z = (t < len) ? accs[et][j] : 0.0f;
      out[idx] = h_e[idx] + z;
    }
  }
}

extern "C" void kernel_launch(void* const* d_in, const int* in_sizes, int n_in,
                              void* d_out, int out_size, void* d_ws,
                              size_t ws_size, hipStream_t stream) {
  const float* h_e = (const float*)d_in[0];
  const float* h_l = (const float*)d_in[1];
  const float* h_i = (const float*)d_in[2];
  const float* h_m = (const float*)d_in[3];
  const float* Wq  = (const float*)d_in[4];
  const float* Wk  = (const float*)d_in[5];
  const float* Wv  = (const float*)d_in[6];
  const int* length = (const int*)d_in[7];
  float* out = (float*)d_out;
  _Float16* ws = (_Float16*)d_ws;   // 35 * 65536 halfs = 4.6 MB used

  swizzle_weights<<<dim3(128, 3), 32, 0, stream>>>(Wq, Wk, Wv, ws);
  proj_kv<<<dim3(B_ * 8), 32, 0, stream>>>(h_l, h_i, h_m, ws, out);
  attention<<<dim3(512), 256, 0, stream>>>(h_e, ws, length, out);
}
